// PASS2S_29231547416763
// MI455X (gfx1250) — matrix operand order repaired and checked by reading, not compiled, not run on hardware
//
#include <hip/hip_runtime.h>
#include <hip/hip_bf16.h>

typedef __bf16 bf16;
typedef __attribute__((ext_vector_type(16))) __bf16 v16bf;
typedef __attribute__((ext_vector_type(8)))  __bf16 v8bf;
typedef __attribute__((ext_vector_type(8)))  float  v8f;

#define NS   8      // streams
#define TENC 12     // encoder timesteps
#define NH   256    // hidden
#define NF   316    // input features
#define FPAD 320    // padded K for encoder input GEMM
#define NB   1024   // batch
#define LMAX 48     // decoder steps
#define G4H  1024   // 4*NH

// ---------- scalar helpers ----------
__device__ __forceinline__ bf16 to_bf16(float f) {
  union { float f; unsigned u; } v; v.f = f;
  unsigned r = v.u + 0x7FFFu + ((v.u >> 16) & 1u);
  union { unsigned short s; bf16 b; } o; o.s = (unsigned short)(r >> 16);
  return o.b;
}
__device__ __forceinline__ float bf2f(bf16 b) {
  union { unsigned short s; bf16 b; } i; i.b = b;
  union { unsigned u; float f; } o; o.u = ((unsigned)i.s) << 16;
  return o.f;
}
__device__ __forceinline__ float sigf(float x) { return 1.0f / (1.0f + __expf(-x)); }

// ---------- WMMA fragment loaders (bf16, wave32) ----------
// A matrix 16x32 (MxK), row-major source [*, lda]:
// lanes 0-15: M=lane, K = k0+0..7 (VGPR0-3) and k0+16..23 (VGPR4-7)
// lanes 16-31: M=lane-16, K = k0+8..15 and k0+24..31
__device__ __forceinline__ v16bf load_a(const bf16* __restrict__ base, int lda,
                                        int m0, int k0, int lane) {
  int row = m0 + (lane & 15);
  int kb  = k0 + ((lane & 16) ? 8 : 0);
  const bf16* p = base + (size_t)row * lda + kb;
  v8bf lo = *(const v8bf*)(p);
  v8bf hi = *(const v8bf*)(p + 16);
  v16bf a;
#pragma unroll
  for (int i = 0; i < 8; ++i) { a[i] = lo[i]; a[i + 8] = hi[i]; }
  return a;
}
// B matrix 32x16 (KxN) sourced from row-major weights W[N,K] (i.e. B = W^T):
// lanes 0-15: N=lane, K=k0+0..7 and k0+16..23 ; lanes 16-31: N=lane-16, K=k0+8..15 and k0+24..31
__device__ __forceinline__ v16bf load_b(const bf16* __restrict__ wbase, int ldw,
                                        int n0, int k0, int lane) {
  int col = n0 + (lane & 15);
  int kb  = k0 + ((lane & 16) ? 8 : 0);
  const bf16* p = wbase + (size_t)col * ldw + kb;
  v8bf lo = *(const v8bf*)(p);
  v8bf hi = *(const v8bf*)(p + 16);
  v16bf b;
#pragma unroll
  for (int i = 0; i < 8; ++i) { b[i] = lo[i]; b[i + 8] = hi[i]; }
  return b;
}
__device__ __forceinline__ v8f wmma_bf16(v16bf a, v16bf b, v8f c) {
  return __builtin_amdgcn_wmma_f32_16x16x32_bf16(false, a, false, b, (short)0, c,
                                                 false, false);
}
__device__ __forceinline__ v8f vzero8() {
  v8f z;
#pragma unroll
  for (int i = 0; i < 8; ++i) z[i] = 0.0f;
  return z;
}

// ---------- conversion / init kernels ----------
__global__ void convx_kernel(const float* __restrict__ x, bf16* __restrict__ xp) {
  long i = (long)blockIdx.x * 256 + threadIdx.x;
  const long tot = (long)NS * TENC * NB * FPAD;
  if (i >= tot) return;
  int f = (int)(i % FPAD);
  long r = i / FPAD;
  int b  = (int)(r % NB);
  int st = (int)(r / NB);                // s*TENC + t
  xp[i] = (f < NF) ? to_bf16(x[((size_t)b * 96 + st) * NF + f]) : to_bf16(0.0f);
}
__global__ void padw_kernel(const float* __restrict__ in, bf16* __restrict__ out,
                            int rows, int kin, int kpad) {
  long i = (long)blockIdx.x * 256 + threadIdx.x;
  long tot = (long)rows * kpad;
  if (i >= tot) return;
  int f = (int)(i % kpad);
  long r = i / kpad;
  out[i] = (f < kin) ? to_bf16(in[r * kin + f]) : to_bf16(0.0f);
}
__global__ void f2bf_kernel(const float* __restrict__ in, bf16* __restrict__ out, long n) {
  long i = (long)blockIdx.x * 256 + threadIdx.x;
  if (i < n) out[i] = to_bf16(in[i]);
}
__global__ void zero_kernel(unsigned int* __restrict__ p, long n) {
  long i = (long)blockIdx.x * 256 + threadIdx.x;
  if (i < n) p[i] = 0u;
}

// ---------- fused encoder LSTM step: gates = x@Wih^T + h@Whh^T + b; h,c update ----
// Wave tile: M=32 (two 16-row A frags share each B frag) x N=16 x 4 gates.
__global__ __launch_bounds__(256) void enc_step_kernel(
    const bf16* __restrict__ xp, int t,
    const bf16* __restrict__ wih, const bf16* __restrict__ whh,
    const float* __restrict__ bih, const float* __restrict__ bhh,
    const bf16* __restrict__ h_in, bf16* __restrict__ h_out,
    float* __restrict__ c_st, bf16* __restrict__ enc_out) {
  int lane = threadIdx.x & 31;
  int wid  = (blockIdx.x << 3) + (threadIdx.x >> 5);   // 0..511
  int s    = blockIdx.y;
  int m0 = (wid >> 4) << 5;   // batch tile (32 rows)
  int n0 = (wid & 15) << 4;   // H tile (16 cols)

  const bf16* Ax = xp   + ((size_t)(s * TENC + t) * NB) * FPAD;
  const bf16* Ah = h_in + (size_t)s * NB * NH;
  const bf16* Wi = wih  + (size_t)s * G4H * FPAD;
  const bf16* Wh = whh  + (size_t)s * G4H * NH;

  v8f acc[4][2];
#pragma unroll
  for (int g = 0; g < 4; ++g) { acc[g][0] = vzero8(); acc[g][1] = vzero8(); }

  for (int k0 = 0; k0 < FPAD; k0 += 32) {
    v16bf a0 = load_a(Ax, FPAD, m0, k0, lane);
    v16bf a1 = load_a(Ax, FPAD, m0 + 16, k0, lane);
#pragma unroll
    for (int g = 0; g < 4; ++g) {
      v16bf bf = load_b(Wi, FPAD, n0 + (g << 8), k0, lane);
      acc[g][0] = wmma_bf16(a0, bf, acc[g][0]);
      acc[g][1] = wmma_bf16(a1, bf, acc[g][1]);
    }
  }
  for (int k0 = 0; k0 < NH; k0 += 32) {
    v16bf a0 = load_a(Ah, NH, m0, k0, lane);
    v16bf a1 = load_a(Ah, NH, m0 + 16, k0, lane);
#pragma unroll
    for (int g = 0; g < 4; ++g) {
      v16bf bf = load_b(Wh, NH, n0 + (g << 8), k0, lane);
      acc[g][0] = wmma_bf16(a0, bf, acc[g][0]);
      acc[g][1] = wmma_bf16(a1, bf, acc[g][1]);
    }
  }

  int col = n0 + (lane & 15);
  float bs[4];
#pragma unroll
  for (int g = 0; g < 4; ++g) {
    int gi = s * G4H + (g << 8) + col;
    bs[g] = bih[gi] + bhh[gi];
  }
#pragma unroll
  for (int half = 0; half < 2; ++half) {
    int rowb = m0 + (half << 4) + ((lane & 16) ? 8 : 0);
#pragma unroll
    for (int r = 0; r < 8; ++r) {
      int b = rowb + r;
      size_t idx = ((size_t)s * NB + b) * NH + col;
      float iv = sigf(acc[0][half][r] + bs[0]);
      float fv = sigf(acc[1][half][r] + bs[1]);
      float gv = tanhf(acc[2][half][r] + bs[2]);
      float ov = sigf(acc[3][half][r] + bs[3]);
      float cn = fv * c_st[idx] + iv * gv;
      float hn = ov * tanhf(cn);
      c_st[idx] = cn;
      bf16 hb = to_bf16(hn);
      h_out[idx] = hb;
      enc_out[((size_t)(s * TENC + t) * NB + b) * NH + col] = hb;
    }
  }
}

// ---------- fused decoder LSTM step (input GEMM degenerates to scalar broadcast) ----
__global__ __launch_bounds__(256) void dec_step_kernel(
    const float* __restrict__ x, const float* __restrict__ tgt, int l,
    const float* __restrict__ wihd, const bf16* __restrict__ whh,
    const float* __restrict__ bih, const float* __restrict__ bhh,
    const bf16* __restrict__ h_in, bf16* __restrict__ h_out,
    float* __restrict__ h_f32, float* __restrict__ c_st) {
  int lane = threadIdx.x & 31;
  int wid  = (blockIdx.x << 3) + (threadIdx.x >> 5);   // 0..511
  int s    = blockIdx.y;
  int m0 = (wid >> 4) << 5;
  int n0 = (wid & 15) << 4;

  const bf16* Ah = h_in + (size_t)s * NB * NH;
  const bf16* Wh = whh  + (size_t)s * G4H * NH;

  v8f acc[4][2];
#pragma unroll
  for (int g = 0; g < 4; ++g) { acc[g][0] = vzero8(); acc[g][1] = vzero8(); }

  for (int k0 = 0; k0 < NH; k0 += 32) {
    v16bf a0 = load_a(Ah, NH, m0, k0, lane);
    v16bf a1 = load_a(Ah, NH, m0 + 16, k0, lane);
#pragma unroll
    for (int g = 0; g < 4; ++g) {
      v16bf bf = load_b(Wh, NH, n0 + (g << 8), k0, lane);
      acc[g][0] = wmma_bf16(a0, bf, acc[g][0]);
      acc[g][1] = wmma_bf16(a1, bf, acc[g][1]);
    }
  }

  int col = n0 + (lane & 15);
  float bs[4], wg[4];
#pragma unroll
  for (int g = 0; g < 4; ++g) {
    int gi = s * G4H + (g << 8) + col;
    bs[g] = bih[gi] + bhh[gi];
    wg[g] = wihd[gi];
  }
#pragma unroll
  for (int half = 0; half < 2; ++half) {
    int rowb = m0 + (half << 4) + ((lane & 16) ? 8 : 0);
#pragma unroll
    for (int r = 0; r < 8; ++r) {
      int b = rowb + r;
      float inp = (l == 0) ? x[((size_t)b * 96 + 95) * NF]
                           : tgt[(size_t)b * LMAX + (l - 1)];
      size_t idx = ((size_t)s * NB + b) * NH + col;
      float iv = sigf(acc[0][half][r] + inp * wg[0] + bs[0]);
      float fv = sigf(acc[1][half][r] + inp * wg[1] + bs[1]);
      float gv = tanhf(acc[2][half][r] + inp * wg[2] + bs[2]);
      float ov = sigf(acc[3][half][r] + inp * wg[3] + bs[3]);
      float cn = fv * c_st[idx] + iv * gv;
      float hn = ov * tanhf(cn);
      c_st[idx]  = cn;
      h_f32[idx] = hn;
      h_out[idx] = to_bf16(hn);
    }
  }
}

// ---------- generic per-stream WMMA GEMM: out = A[M,K](bf16) @ W[N,K]^T + bias ----
// Wave tile: M=32 x N=16 (B frag reused by two A frags).
__global__ __launch_bounds__(256) void gemm_bias_kernel(
    const bf16* __restrict__ A, size_t sAs, int lda,
    const bf16* __restrict__ W, size_t sWs, int ldw,
    const float* __restrict__ bias, int sBs,
    float* __restrict__ outF, size_t sOFs, int ldo,
    bf16* __restrict__ outB, size_t sOBs, int ldob,
    int mtiles32, int ntiles, int K) {
  int lane = threadIdx.x & 31;
  int wid  = (blockIdx.x << 3) + (threadIdx.x >> 5);
  if (wid >= mtiles32 * ntiles) return;        // wave-uniform
  int s  = blockIdx.y;
  int m0 = (wid / ntiles) << 5;
  int n0 = (wid % ntiles) << 4;
  const bf16* As = A + (size_t)s * sAs;
  const bf16* Ws = W + (size_t)s * sWs;

  v8f acc0 = vzero8(), acc1 = vzero8();
  for (int k0 = 0; k0 < K; k0 += 32) {
    v16bf bf = load_b(Ws, ldw, n0, k0, lane);
    acc0 = wmma_bf16(load_a(As, lda, m0, k0, lane), bf, acc0);
    acc1 = wmma_bf16(load_a(As, lda, m0 + 16, k0, lane), bf, acc1);
  }

  int col = n0 + (lane & 15);
  float bv = bias ? bias[s * sBs + col] : 0.0f;
#pragma unroll
  for (int half = 0; half < 2; ++half) {
    int rowb = m0 + (half << 4) + ((lane & 16) ? 8 : 0);
#pragma unroll
    for (int r = 0; r < 8; ++r) {
      int row = rowb + r;
      float v = (half ? acc1[r] : acc0[r]) + bv;
      if (outF) outF[(size_t)s * sOFs + (size_t)row * ldo + col] = v;
      if (outB) outB[(size_t)s * sOBs + (size_t)row * ldob + col] = to_bf16(v);
    }
  }
}

// ---------- attention core: scores, softmax over TENC, ctx = attn@V ----------
__global__ __launch_bounds__(256) void attn_kernel(
    const float* __restrict__ q, const bf16* __restrict__ kv,
    bf16* __restrict__ ctx) {
  int idx = blockIdx.x * 256 + threadIdx.x;   // 0 .. NS*NB-1
  int s = idx >> 10, b = idx & 1023;
  const float* qp = q + ((size_t)s * NB + b) * NH;
  const bf16* kb0 = kv + ((size_t)s * TENC * NB + b) * (2 * NH);
  float sc[TENC];
#pragma unroll 1
  for (int t = 0; t < TENC; ++t) {
    const bf16* kp = kb0 + (size_t)t * NB * (2 * NH);
    float d = 0.0f;
    for (int e = 0; e < NH; ++e) d += qp[e] * bf2f(kp[e]);
    sc[t] = d * 0.0625f;   // 1/sqrt(256)
  }
  float m = sc[0];
#pragma unroll
  for (int t = 1; t < TENC; ++t) m = fmaxf(m, sc[t]);
  float ssum = 0.0f;
#pragma unroll
  for (int t = 0; t < TENC; ++t) { sc[t] = __expf(sc[t] - m); ssum += sc[t]; }
  float inv = 1.0f / ssum;
#pragma unroll
  for (int t = 0; t < TENC; ++t) sc[t] *= inv;
  bf16* cp = ctx + ((size_t)s * NB + b) * NH;
#pragma unroll 1
  for (int e = 0; e < NH; ++e) {
    float a = 0.0f;
#pragma unroll
    for (int t = 0; t < TENC; ++t)
      a += sc[t] * bf2f(kb0[(size_t)t * NB * (2 * NH) + NH + e]);
    cp[e] = to_bf16(a);
  }
}

// ---------- final: out_fc(2H->1) per stream, then fc(8->1) ----------
__global__ void finalize_kernel(const float* __restrict__ hf,
                                const float* __restrict__ ctxp,
                                const float* __restrict__ wfco,
                                const float* __restrict__ bfco,
                                const float* __restrict__ wfc,
                                const float* __restrict__ bfc,
                                float* __restrict__ y, int l) {
  int b = blockIdx.x * 256 + threadIdx.x;
  if (b >= NB) return;
  float acc = bfc[0];
  for (int s = 0; s < NS; ++s) {
    const float* hp = hf   + ((size_t)s * NB + b) * NH;
    const float* cp = ctxp + ((size_t)s * NB + b) * NH;
    const float* w  = wfco + (size_t)s * 2 * NH;
    float d = bfco[s];
    for (int j = 0; j < NH; ++j) d += hp[j] * w[j] + cp[j] * w[NH + j];
    acc += d * wfc[s];
  }
  y[(size_t)b * LMAX + l] = acc;
}

// ---------------------------------------------------------------------------
extern "C" void kernel_launch(void* const* d_in, const int* in_sizes, int n_in,
                              void* d_out, int out_size, void* d_ws, size_t ws_size,
                              hipStream_t stream) {
  const float* x     = (const float*)d_in[0];
  const float* tgt   = (const float*)d_in[1];
  const float* Wihe  = (const float*)d_in[2];
  const float* Whhe  = (const float*)d_in[3];
  const float* bihe  = (const float*)d_in[4];
  const float* bhhe  = (const float*)d_in[5];
  const float* Wihd  = (const float*)d_in[6];
  const float* Whhd  = (const float*)d_in[7];
  const float* bihd  = (const float*)d_in[8];
  const float* bhhd  = (const float*)d_in[9];
  const float* Wattn = (const float*)d_in[10];
  const float* battn = (const float*)d_in[11];
  const float* Wout  = (const float*)d_in[12];
  const float* bout  = (const float*)d_in[13];
  const float* Wfco  = (const float*)d_in[14];
  const float* bfco  = (const float*)d_in[15];
  const float* Wfc   = (const float*)d_in[16];
  const float* bfc   = (const float*)d_in[17];
  float* y = (float*)d_out;

  // ---- workspace layout ----
  constexpr size_t SZ_XP   = (size_t)NS * TENC * NB * FPAD * 2;   // 60 MB
  constexpr size_t SZ_WIHE = (size_t)NS * G4H * FPAD * 2;         // 5 MB
  constexpr size_t SZ_WHH  = (size_t)NS * G4H * NH * 2;           // 4 MB
  constexpr size_t SZ_WATT = (size_t)NS * 3 * NH * NH * 2;        // 3 MB
  constexpr size_t SZ_WOUT = (size_t)NS * NH * NH * 2;            // 1 MB
  constexpr size_t SZ_H    = (size_t)NS * NB * NH * 2;            // 4 MB (bf16)
  constexpr size_t SZ_HF   = (size_t)NS * NB * NH * 4;            // 8 MB (f32)
  constexpr size_t SZ_ENC  = (size_t)NS * TENC * NB * NH * 2;     // 48 MB
  constexpr size_t SZ_KV   = (size_t)NS * TENC * NB * 2 * NH * 2; // 96 MB

  char* ws = (char*)d_ws;
  bf16*  xp    = (bf16*)ws;                     ws += SZ_XP;
  bf16*  wihe  = (bf16*)ws;                     ws += SZ_WIHE;
  bf16*  whhe  = (bf16*)ws;                     ws += SZ_WHH;
  bf16*  whhd  = (bf16*)ws;                     ws += SZ_WHH;
  bf16*  wattn = (bf16*)ws;                     ws += SZ_WATT;
  bf16*  wout  = (bf16*)ws;                     ws += SZ_WOUT;
  bf16*  hA    = (bf16*)ws;                     ws += SZ_H;
  bf16*  hB    = (bf16*)ws;                     ws += SZ_H;
  float* c_st  = (float*)ws;                    ws += SZ_HF;
  float* h_f32 = (float*)ws;                    ws += SZ_HF;
  bf16*  enc   = (bf16*)ws;                     ws += SZ_ENC;
  bf16*  kvb   = (bf16*)ws;                     ws += SZ_KV;
  float* qf    = (float*)ws;                    ws += SZ_HF;
  bf16*  ctxb  = (bf16*)ws;                     ws += SZ_H;
  float* ctxp  = (float*)ws;                    ws += SZ_HF;
  (void)ws_size; (void)in_sizes; (void)n_in; (void)out_size;

  // ---- one-shot conversions / init ----
  {
    long n = (long)NS * TENC * NB * FPAD;
    convx_kernel<<<(unsigned)((n + 255) / 256), 256, 0, stream>>>(x, xp);
  }
  {
    long n = (long)NS * G4H * FPAD;
    padw_kernel<<<(unsigned)((n + 255) / 256), 256, 0, stream>>>(Wihe, wihe, NS * G4H, NF, FPAD);
  }
  {
    long n = (long)NS * G4H * NH;
    f2bf_kernel<<<(unsigned)((n + 255) / 256), 256, 0, stream>>>(Whhe, whhe, n);
    f2bf_kernel<<<(unsigned)((n + 255) / 256), 256, 0, stream>>>(Whhd, whhd, n);
  }
  {
    long n = (long)NS * 3 * NH * NH;
    f2bf_kernel<<<(unsigned)((n + 255) / 256), 256, 0, stream>>>(Wattn, wattn, n);
  }
  {
    long n = (long)NS * NH * NH;
    f2bf_kernel<<<(unsigned)((n + 255) / 256), 256, 0, stream>>>(Wout, wout, n);
  }
  {
    long n = (long)(SZ_H / 4);
    zero_kernel<<<(unsigned)((n + 255) / 256), 256, 0, stream>>>((unsigned int*)hA, n);
    n = (long)(SZ_HF / 4);
    zero_kernel<<<(unsigned)((n + 255) / 256), 256, 0, stream>>>((unsigned int*)c_st, n);
  }

  // ---- encoder scan ----  (512 waves/stream -> 64 blocks x 8 streams)
  bf16* hin = hA;
  bf16* hout = hB;
  for (int t = 0; t < TENC; ++t) {
    enc_step_kernel<<<dim3(64, NS), 256, 0, stream>>>(xp, t, wihe, whhe, bihe, bhhe,
                                                      hin, hout, c_st, enc);
    bf16* tmp = hin; hin = hout; hout = tmp;   // hin = newest h
  }

  // ---- K/V projection: [T*B,256] @ W[k;v][512,256]^T + b -> kv bf16 [S,T,B,512] ----
  // mtiles32 = 12288/32 = 384, ntiles = 32 -> 12288 waves -> 1536 blocks
  gemm_bias_kernel<<<dim3(1536, NS), 256, 0, stream>>>(
      enc, (size_t)TENC * NB * NH, NH,
      wattn + (size_t)NH * NH, (size_t)3 * NH * NH, NH,
      battn + NH, 3 * NH,
      nullptr, 0, 0,
      kvb, (size_t)TENC * NB * 2 * NH, 2 * NH,
      (TENC * NB) / 32, (2 * NH) / 16, NH);

  // ---- decoder scan ----
  for (int l = 0; l < LMAX; ++l) {
    dec_step_kernel<<<dim3(64, NS), 256, 0, stream>>>(x, tgt, l, Wihd, whhd, bihd, bhhd,
                                                      hin, hout, h_f32, c_st);
    bf16* tmp = hin; hin = hout; hout = tmp;   // hin = h for this step

    // q = h @ Wq^T + bq  (f32 out) : 32x16 tiles -> 512 waves -> 64 blocks
    gemm_bias_kernel<<<dim3(64, NS), 256, 0, stream>>>(
        hin, (size_t)NB * NH, NH,
        wattn, (size_t)3 * NH * NH, NH,
        battn, 3 * NH,
        qf, (size_t)NB * NH, NH,
        nullptr, 0, 0,
        NB / 32, NH / 16, NH);

    attn_kernel<<<(NS * NB) / 256, 256, 0, stream>>>(qf, kvb, ctxb);

    // ctx_proj = ctx @ Wout^T + bout (f32 out)
    gemm_bias_kernel<<<dim3(64, NS), 256, 0, stream>>>(
        ctxb, (size_t)NB * NH, NH,
        wout, (size_t)NH * NH, NH,
        bout, NH,
        ctxp, (size_t)NB * NH, NH,
        nullptr, 0, 0,
        NB / 32, NH / 16, NH);

    finalize_kernel<<<NB / 256, 256, 0, stream>>>(h_f32, ctxp, Wfco, bfco, Wfc, bfc, y, l);
  }
}
